// Attention_20255065768401
// MI455X (gfx1250) — matrix operand order repaired and checked by reading, not compiled, not run on hardware
//
#include <hip/hip_runtime.h>
#include <hip/hip_bf16.h>

#define D_MODEL 1024
#define N_HEAD  16
#define D_HEAD  64
#define SEQ     2048
#define BATCH   2
#define M_ROWS  (BATCH * SEQ)   // 4096

typedef __attribute__((ext_vector_type(16))) _Float16 v16h;
typedef __attribute__((ext_vector_type(8)))  _Float16 v8h;
typedef __attribute__((ext_vector_type(8)))  float    v8f;

#define WMMA_F16(A, B, C) \
  __builtin_amdgcn_wmma_f32_16x16x32_f16(false, (A), false, (B), (short)0, (C), false, false)

static __device__ __forceinline__ v8f zero8() {
  v8f z;
#pragma unroll
  for (int i = 0; i < 8; ++i) z[i] = 0.0f;
  return z;
}

// ---- CDNA5 async global->LDS copy (ASYNCcnt path), 16B per lane ------------
static __device__ __forceinline__ void async_copy_b128(unsigned lds_byte_off,
                                                       const void* gptr) {
  asm volatile("global_load_async_to_lds_b128 %0, %1, off"
               :: "v"(lds_byte_off), "v"(gptr)
               : "memory");
}
static __device__ __forceinline__ void wait_async_zero() {
  asm volatile("s_wait_asynccnt 0x0" ::: "memory");
}

// A fragment (16x32 f16, M x K): lane L<16 holds row M=L, k = {0..7, 16..23};
// lane L>=16 holds row M=L-16, k = {8..15, 24..31}.  Row-major src, ld in elems.
static __device__ __forceinline__ v16h load_a_frag_g(const _Float16* __restrict__ base,
                                                     int ld, int lane) {
  const int m  = lane & 15;
  const int kh = (lane >> 4) << 3;            // 0 or 8
  const _Float16* p = base + (size_t)m * ld + kh;
  v8h lo = *(const v8h*)(p);                  // k = kh .. kh+7
  v8h hi = *(const v8h*)(p + 16);             // k = 16+kh .. 16+kh+7
  v16h a;
#pragma unroll
  for (int i = 0; i < 8; ++i) { a[i] = lo[i]; a[i + 8] = hi[i]; }
  return a;
}

// B fragment (32x16 f16, K x N) from row-major [N, K] storage:
// lane L<16 holds col N=L, k = {0..7, 16..23}; lane L>=16 holds col N=L-16, k = {8..15, 24..31}.
static __device__ __forceinline__ v16h load_b_frag_g(const _Float16* __restrict__ bmat,
                                                     int ld, int lane) {
  const int n  = lane & 15;
  const int kb = (lane >> 4) << 3;            // 0 or 8
  const _Float16* p = bmat + (size_t)n * ld + kb;
  v8h lo = *(const v8h*)(p);
  v8h hi = *(const v8h*)(p + 16);
  v16h b;
#pragma unroll
  for (int i = 0; i < 8; ++i) { b[i] = lo[i]; b[i + 8] = hi[i]; }
  return b;
}

// ---------------------------------------------------------------- utilities
__global__ void cvt_f32_to_f16(const float* __restrict__ src,
                               _Float16* __restrict__ dst, int n) {
  int i = blockIdx.x * blockDim.x + threadIdx.x;
  if (i < n) dst[i] = (_Float16)src[i];
}

__global__ void rope_table_kernel(float* __restrict__ cosT, float* __restrict__ sinT) {
  int i = blockIdx.x * blockDim.x + threadIdx.x;
  if (i >= SEQ * 32) return;
  int s = i >> 5, j = i & 31;
  float freq = powf(10000.0f, -(float)j / 32.0f);
  float ang  = (float)s * freq;
  cosT[i] = cosf(ang);
  sinT[i] = sinf(ang);
}

// ------------------------------------------------ QKV GEMM + RoPE + scatter
// x_h [4096,1024] f16 row-major, w_h [3072,1024] f16 row-major ([N,K]).
// Block stages its 16x1024 A panel (32KB, contiguous) into LDS with async
// copies; each wave computes a 16(M) x 64(N) tile == one (which, head) slice.
__global__ void qkv_rope_kernel(const _Float16* __restrict__ xh,
                                const _Float16* __restrict__ wh,
                                const float* __restrict__ cosT,
                                const float* __restrict__ sinT,
                                _Float16* __restrict__ qb,   // [B,H,S,64]
                                _Float16* __restrict__ kb,   // [B,H,S,64]
                                _Float16* __restrict__ vt) { // [B,H,64,S]
  __shared__ __align__(16) _Float16 sA[16 * D_MODEL];

  const int lane  = threadIdx.x & 31;
  const int wave  = threadIdx.x >> 5;
  const int mtile = blockIdx.x;                       // 16 rows of B*S
  const int n0    = blockIdx.y * 512 + wave * 64;     // 64-wide N tile

  // Async-stage the A panel: 2048 x 16B chunks over 256 threads.
  {
    const _Float16* Ag = xh + (size_t)mtile * 16 * D_MODEL;
#pragma unroll
    for (int i = 0; i < 8; ++i) {
      const int c = threadIdx.x + i * 256;            // 16-byte chunk id
      const unsigned lofs = (unsigned)(uintptr_t)(const void*)&sA[c * 8];
      async_copy_b128(lofs, (const void*)(Ag + c * 8));
    }
    wait_async_zero();
    __syncthreads();
  }

  const int mA  = lane & 15;
  const int khA = (lane >> 4) << 3;

  v8f acc0 = zero8(), acc1 = zero8(), acc2 = zero8(), acc3 = zero8();
  for (int k0 = 0; k0 < D_MODEL; k0 += 32) {
    v16h a;
    {
      v8h lo = *(const v8h*)&sA[mA * D_MODEL + k0 + khA];
      v8h hi = *(const v8h*)&sA[mA * D_MODEL + k0 + khA + 16];
#pragma unroll
      for (int i = 0; i < 8; ++i) { a[i] = lo[i]; a[i + 8] = hi[i]; }
    }
    v16h b0 = load_b_frag_g(wh + (size_t)(n0 +  0) * D_MODEL + k0, D_MODEL, lane);
    v16h b1 = load_b_frag_g(wh + (size_t)(n0 + 16) * D_MODEL + k0, D_MODEL, lane);
    v16h b2 = load_b_frag_g(wh + (size_t)(n0 + 32) * D_MODEL + k0, D_MODEL, lane);
    v16h b3 = load_b_frag_g(wh + (size_t)(n0 + 48) * D_MODEL + k0, D_MODEL, lane);
    acc0 = WMMA_F16(a, b0, acc0);
    acc1 = WMMA_F16(a, b1, acc1);
    acc2 = WMMA_F16(a, b2, acc2);
    acc3 = WMMA_F16(a, b3, acc3);
  }

  const int three = n0 >> 10;           // 0=q, 1=k, 2=v
  const int h     = (n0 >> 6) & 15;
  const int n     = lane & 15;          // column within 16-wide subtile
  const int mhi   = (lane >> 4) << 3;   // 0 or 8

  if (three < 2) {
    _Float16* dstbase = (three == 0) ? qb : kb;
#pragma unroll
    for (int r = 0; r < 8; ++r) {
      int grow = mtile * 16 + r + mhi;
      int bi = grow >> 11, s = grow & (SEQ - 1);
      float c0 = cosT[s * 32 + n],      s0 = sinT[s * 32 + n];
      float c1 = cosT[s * 32 + 16 + n], s1 = sinT[s * 32 + 16 + n];
      // rotate_half: d<32 -> x[d]*cos - x[d+32]*sin ; d>=32 -> x[d]*cos + x[d-32]*sin
      float o0 = acc0[r] * c0 - acc2[r] * s0;   // d = n
      float o1 = acc1[r] * c1 - acc3[r] * s1;   // d = 16+n
      float o2 = acc2[r] * c0 + acc0[r] * s0;   // d = 32+n
      float o3 = acc3[r] * c1 + acc1[r] * s1;   // d = 48+n
      _Float16* dst = dstbase + (((size_t)(bi * N_HEAD + h) * SEQ + s) * D_HEAD);
      dst[n]      = (_Float16)o0;
      dst[16 + n] = (_Float16)o1;
      dst[32 + n] = (_Float16)o2;
      dst[48 + n] = (_Float16)o3;
    }
  } else {
#pragma unroll
    for (int r = 0; r < 8; ++r) {
      int grow = mtile * 16 + r + mhi;
      int bi = grow >> 11, s = grow & (SEQ - 1);
      _Float16* dst = vt + (size_t)(bi * N_HEAD + h) * D_HEAD * SEQ + s;
      dst[(size_t)(n)      * SEQ] = (_Float16)acc0[r];
      dst[(size_t)(16 + n) * SEQ] = (_Float16)acc1[r];
      dst[(size_t)(32 + n) * SEQ] = (_Float16)acc2[r];
      dst[(size_t)(48 + n) * SEQ] = (_Float16)acc3[r];
    }
  }
}

// ----------------------------------------------------- causal flash attention
// One wave per 16-row q tile; k processed in chunks of 32 with online softmax.
__global__ void flash_attn_kernel(const _Float16* __restrict__ qb,
                                  const _Float16* __restrict__ kb,
                                  const _Float16* __restrict__ vt,
                                  _Float16* __restrict__ zh) { // [B,S,1024]
  __shared__ __align__(16) _Float16 plds[16 * 32];
  const int lane  = threadIdx.x & 31;
  const int qt    = blockIdx.x;
  const int head  = blockIdx.y;
  const int batch = blockIdx.z;
  const int bh    = batch * N_HEAD + head;
  const int qrow0 = qt * 16;

  const _Float16* Qbase = qb + ((size_t)bh * SEQ + qrow0) * D_HEAD;
  v16h aq0 = load_a_frag_g(Qbase,      D_HEAD, lane);   // d 0..31
  v16h aq1 = load_a_frag_g(Qbase + 32, D_HEAD, lane);   // d 32..63

  v8f acc0 = zero8(), acc1 = zero8(), acc2 = zero8(), acc3 = zero8();
  float mrun[8], lrun[8];
#pragma unroll
  for (int r = 0; r < 8; ++r) { mrun[r] = -__builtin_inff(); lrun[r] = 0.0f; }

  const int n   = lane & 15;
  const int mhi = (lane >> 4) << 3;
  const int nj  = (qrow0 + 15) / 32 + 1;

  for (int j = 0; j < nj; ++j) {
    const int kbase = j * 32;
    const _Float16* K0 = kb + ((size_t)bh * SEQ + kbase) * D_HEAD;

    // scores: 16(q) x 32(k), reduction over d=64
    v8f s0 = zero8(), s1 = zero8();
    {
      v16h bk;
      bk = load_b_frag_g(K0,                    D_HEAD, lane); s0 = WMMA_F16(aq0, bk, s0);
      bk = load_b_frag_g(K0 + 32,               D_HEAD, lane); s0 = WMMA_F16(aq1, bk, s0);
      bk = load_b_frag_g(K0 + 16 * D_HEAD,      D_HEAD, lane); s1 = WMMA_F16(aq0, bk, s1);
      bk = load_b_frag_g(K0 + 16 * D_HEAD + 32, D_HEAD, lane); s1 = WMMA_F16(aq1, bk, s1);
    }

#pragma unroll
    for (int r = 0; r < 8; ++r) {
      const int qg = qrow0 + r + mhi;
      float v0 = s0[r] * 0.125f;
      float v1 = s1[r] * 0.125f;
      if (kbase + n > qg)      v0 = -__builtin_inff();
      if (kbase + 16 + n > qg) v1 = -__builtin_inff();
      float rm = fmaxf(v0, v1);
      rm = fmaxf(rm, __shfl_xor(rm, 1, 32));
      rm = fmaxf(rm, __shfl_xor(rm, 2, 32));
      rm = fmaxf(rm, __shfl_xor(rm, 4, 32));
      rm = fmaxf(rm, __shfl_xor(rm, 8, 32));
      const float mnew = fmaxf(mrun[r], rm);
      const float corr = __expf(mrun[r] - mnew);
      const float p0 = __expf(v0 - mnew);
      const float p1 = __expf(v1 - mnew);
      float rs = p0 + p1;
      rs += __shfl_xor(rs, 1, 32);
      rs += __shfl_xor(rs, 2, 32);
      rs += __shfl_xor(rs, 4, 32);
      rs += __shfl_xor(rs, 8, 32);
      lrun[r] = lrun[r] * corr + rs;
      mrun[r] = mnew;
      acc0[r] *= corr; acc1[r] *= corr; acc2[r] *= corr; acc3[r] *= corr;
      const int m = r + mhi;
      plds[m * 32 + n]      = (_Float16)p0;
      plds[m * 32 + 16 + n] = (_Float16)p1;
    }
    __syncthreads();

    // P tile (C layout) -> A-fragment via LDS
    v16h pa;
    {
      const int m  = lane & 15;
      const int kh = (lane >> 4) << 3;
      v8h lo = *(const v8h*)&plds[m * 32 + kh];
      v8h hi = *(const v8h*)&plds[m * 32 + kh + 16];
#pragma unroll
      for (int i = 0; i < 8; ++i) { pa[i] = lo[i]; pa[i + 8] = hi[i]; }
    }
    __syncthreads();

    const _Float16* V0 = vt + (size_t)bh * D_HEAD * SEQ + kbase;  // [d, s]
    v16h vb;
    vb = load_b_frag_g(V0,                      SEQ, lane); acc0 = WMMA_F16(pa, vb, acc0);
    vb = load_b_frag_g(V0 + (size_t)16 * SEQ,   SEQ, lane); acc1 = WMMA_F16(pa, vb, acc1);
    vb = load_b_frag_g(V0 + (size_t)32 * SEQ,   SEQ, lane); acc2 = WMMA_F16(pa, vb, acc2);
    vb = load_b_frag_g(V0 + (size_t)48 * SEQ,   SEQ, lane); acc3 = WMMA_F16(pa, vb, acc3);
  }

#pragma unroll
  for (int r = 0; r < 8; ++r) {
    const int s = qrow0 + r + mhi;
    const float inv = 1.0f / lrun[r];
    _Float16* dst = zh + ((size_t)batch * SEQ + s) * D_MODEL + head * D_HEAD;
    dst[n]      = (_Float16)(acc0[r] * inv);
    dst[16 + n] = (_Float16)(acc1[r] * inv);
    dst[32 + n] = (_Float16)(acc2[r] * inv);
    dst[48 + n] = (_Float16)(acc3[r] * inv);
  }
}

// ---------------------------------------------------------- output projection
__global__ void out_proj_kernel(const _Float16* __restrict__ zh,
                                const _Float16* __restrict__ woh,  // [N=1024, K=1024]
                                float* __restrict__ out) {
  __shared__ __align__(16) _Float16 sA[16 * D_MODEL];

  const int lane  = threadIdx.x & 31;
  const int wave  = threadIdx.x >> 5;
  const int mtile = blockIdx.x;
  const int n0    = blockIdx.y * 512 + wave * 64;

  {
    const _Float16* Ag = zh + (size_t)mtile * 16 * D_MODEL;
#pragma unroll
    for (int i = 0; i < 8; ++i) {
      const int c = threadIdx.x + i * 256;
      const unsigned lofs = (unsigned)(uintptr_t)(const void*)&sA[c * 8];
      async_copy_b128(lofs, (const void*)(Ag + c * 8));
    }
    wait_async_zero();
    __syncthreads();
  }

  const int mA  = lane & 15;
  const int khA = (lane >> 4) << 3;

  v8f acc0 = zero8(), acc1 = zero8(), acc2 = zero8(), acc3 = zero8();
  for (int k0 = 0; k0 < D_MODEL; k0 += 32) {
    v16h a;
    {
      v8h lo = *(const v8h*)&sA[mA * D_MODEL + k0 + khA];
      v8h hi = *(const v8h*)&sA[mA * D_MODEL + k0 + khA + 16];
#pragma unroll
      for (int i = 0; i < 8; ++i) { a[i] = lo[i]; a[i + 8] = hi[i]; }
    }
    v16h b0 = load_b_frag_g(woh + (size_t)(n0 +  0) * D_MODEL + k0, D_MODEL, lane);
    v16h b1 = load_b_frag_g(woh + (size_t)(n0 + 16) * D_MODEL + k0, D_MODEL, lane);
    v16h b2 = load_b_frag_g(woh + (size_t)(n0 + 32) * D_MODEL + k0, D_MODEL, lane);
    v16h b3 = load_b_frag_g(woh + (size_t)(n0 + 48) * D_MODEL + k0, D_MODEL, lane);
    acc0 = WMMA_F16(a, b0, acc0);
    acc1 = WMMA_F16(a, b1, acc1);
    acc2 = WMMA_F16(a, b2, acc2);
    acc3 = WMMA_F16(a, b3, acc3);
  }

  const int n   = lane & 15;
  const int mhi = (lane >> 4) << 3;
#pragma unroll
  for (int r = 0; r < 8; ++r) {
    const size_t grow = (size_t)(mtile * 16 + r + mhi);
    float* dst = out + grow * D_MODEL + n0;
    dst[n]      = acc0[r];
    dst[16 + n] = acc1[r];
    dst[32 + n] = acc2[r];
    dst[48 + n] = acc3[r];
  }
}

// ---------------------------------------------------------------------- launch
extern "C" void kernel_launch(void* const* d_in, const int* in_sizes, int n_in,
                              void* d_out, int out_size, void* d_ws, size_t ws_size,
                              hipStream_t stream) {
  (void)in_sizes; (void)n_in; (void)out_size; (void)ws_size;
  const float* x     = (const float*)d_in[0];   // [2, 2048, 1024]
  const float* w_qkv = (const float*)d_in[1];   // [3072, 1024]
  const float* w_o   = (const float*)d_in[2];   // [1024, 1024]
  float* out = (float*)d_out;

  char* ws = (char*)d_ws;
  size_t off = 0;
  auto alloc = [&](size_t bytes) -> void* {
    void* p = ws + off;
    off += (bytes + 255) & ~(size_t)255;
    return p;
  };
  _Float16* xh    = (_Float16*)alloc((size_t)M_ROWS * D_MODEL * 2);
  _Float16* wqkvh = (_Float16*)alloc((size_t)3 * D_MODEL * D_MODEL * 2);
  _Float16* woh   = (_Float16*)alloc((size_t)D_MODEL * D_MODEL * 2);
  float*    cosT  = (float*)alloc((size_t)SEQ * 32 * 4);
  float*    sinT  = (float*)alloc((size_t)SEQ * 32 * 4);
  _Float16* qb    = (_Float16*)alloc((size_t)BATCH * N_HEAD * SEQ * D_HEAD * 2);
  _Float16* kb    = (_Float16*)alloc((size_t)BATCH * N_HEAD * SEQ * D_HEAD * 2);
  _Float16* vt    = (_Float16*)alloc((size_t)BATCH * N_HEAD * D_HEAD * SEQ * 2);
  _Float16* zh    = (_Float16*)alloc((size_t)M_ROWS * D_MODEL * 2);

  {
    int n = M_ROWS * D_MODEL;
    cvt_f32_to_f16<<<(n + 255) / 256, 256, 0, stream>>>(x, xh, n);
    n = 3 * D_MODEL * D_MODEL;
    cvt_f32_to_f16<<<(n + 255) / 256, 256, 0, stream>>>(w_qkv, wqkvh, n);
    n = D_MODEL * D_MODEL;
    cvt_f32_to_f16<<<(n + 255) / 256, 256, 0, stream>>>(w_o, woh, n);
    n = SEQ * 32;
    rope_table_kernel<<<(n + 255) / 256, 256, 0, stream>>>(cosT, sinT);
  }

  // QKV GEMM + RoPE: grid = (4096/16, 3072/512), block = 8 waves
  qkv_rope_kernel<<<dim3(M_ROWS / 16, 6), 256, 0, stream>>>(xh, wqkvh, cosT, sinT,
                                                            qb, kb, vt);
  // flash attention: one wave per 16-row q tile
  flash_attn_kernel<<<dim3(SEQ / 16, N_HEAD, BATCH), 32, 0, stream>>>(qb, kb, vt, zh);
  // output projection: grid = (4096/16, 1024/512)
  out_proj_kernel<<<dim3(M_ROWS / 16, 2), 256, 0, stream>>>(zh, woh, out);
}
